// QNN_A_21285857919116
// MI455X (gfx1250) — matrix operand order repaired and checked by reading, not compiled, not run on hardware
//
#include <hip/hip_runtime.h>
#include <math.h>

typedef __attribute__((ext_vector_type(16))) _Float16 v16h;
typedef __attribute__((ext_vector_type(8)))  float    v8f;

#define BATCH   8192
#define PIXELS  784
#define WIRES   10
#define DIM     1024      // 2^WIRES
#define QDEPTH  20
#define NTILES  49        // 784 / 16
#define KTILES  32        // 1024 / 32
#define MTILES  512       // 8192 / 16

// ---------------------------------------------------------------------------
// Kernel 1: linear layer -> angles -> RY product-state amplitudes, written
// directly in WMMA A-fragment (16x32 f16) layout.  One wave per batch row.
// A[b][k] = prod_w ( bit_{9-w}(k) ? sin(th_w/2) : cos(th_w/2) )
//         = hi[k>>5] * lo[k&31]
// ---------------------------------------------------------------------------
__global__ __launch_bounds__(256) void k_embed(const float* __restrict__ x,
                                               const float* __restrict__ W,
                                               const float* __restrict__ bias,
                                               _Float16* __restrict__ Apack)
{
    const int wave = threadIdx.x >> 5;
    const int lane = threadIdx.x & 31;
    const int b    = blockIdx.x * 8 + wave;

    const float* xr = x + (size_t)b * PIXELS;
    float acc[WIRES];
#pragma unroll
    for (int w = 0; w < WIRES; ++w) acc[w] = 0.0f;

    for (int p = lane; p < PIXELS; p += 32) {
        const float xv = xr[p];
#pragma unroll
        for (int w = 0; w < WIRES; ++w)
            acc[w] += xv * W[w * PIXELS + p];
    }

    float cw[WIRES], sw[WIRES];
#pragma unroll
    for (int w = 0; w < WIRES; ++w) {
#pragma unroll
        for (int off = 16; off > 0; off >>= 1)
            acc[w] += __shfl_xor(acc[w], off, 32);
        const float half_ang = 0.5f * (acc[w] + bias[w]);
        __sincosf(half_ang, &sw[w], &cw[w]);
    }

    // hi table over wires 0..4 (bits 9..5), lo table over wires 5..9 (bits 4..0)
    float hiv = 1.0f, lov = 1.0f;
#pragma unroll
    for (int w = 0; w < 5; ++w)
        hiv *= ((lane >> (4 - w)) & 1) ? sw[w] : cw[w];
#pragma unroll
    for (int w = 5; w < WIRES; ++w)
        lov *= ((lane >> (9 - w)) & 1) ? sw[w] : cw[w];

    // Write A in fragment layout.  This lane owns k = kk*32 + lane, so within
    // the K-tile k' == lane:  h = (k'>>3)&1, e = (k'&7) + 8*(k'>>4),
    // fragment lane = (b&15) + 16h.
    const int m   = b & 15;
    const int mt  = b >> 4;
    const int h2  = (lane >> 3) & 1;
    const int e   = (lane & 7) + ((lane >> 4) << 3);
    const int fl  = m + (h2 << 4);
    _Float16* base = Apack + (((size_t)mt * KTILES) * 32 + fl) * 16 + e;

    for (int kk = 0; kk < 32; ++kk) {
        const float val = __shfl(hiv, kk, 32) * lov;
        base[(size_t)kk * 512] = (_Float16)val;   // advance one K-tile (32 lanes * 16 halves)
    }
}

// ---------------------------------------------------------------------------
// Kernel 2: build the fixed circuit unitary.  Block j simulates the 20-layer
// Rot+CNOT circuit on basis state |j>, state vector (1024 complex) in LDS.
// Result row:  B[j][n] = U[n][j]  (so C = A @ B is the full contraction),
// written in WMMA B-fragment (32x16 f16) layout, columns n < 784 only.
// ---------------------------------------------------------------------------
__global__ __launch_bounds__(512) void k_unitary(const float* __restrict__ qw,
                                                 _Float16* __restrict__ Bre,
                                                 _Float16* __restrict__ Bim)
{
    __shared__ float sre[DIM];
    __shared__ float sim_[DIM];
    __shared__ float g[QDEPTH * WIRES * 8];   // m00r,m00i,m01r,m01i,m10r,m10i,m11r,m11i

    const int t = threadIdx.x;
    const int j = blockIdx.x;

    // Precompute all Rot gate matrices once (200 gates).
    if (t < QDEPTH * WIRES) {
        const float phi = qw[t * 3 + 0];
        const float th  = qw[t * 3 + 1];
        const float om  = qw[t * 3 + 2];
        float c, s, cp, sp, cm, sm;
        __sincosf(0.5f * th, &s, &c);
        __sincosf(0.5f * (phi + om), &sp, &cp);
        __sincosf(0.5f * (phi - om), &sm, &cm);
        float* gg = g + t * 8;
        gg[0] =  cp * c;  gg[1] = -sp * c;   // m00 = e^{-i(phi+om)/2} cos
        gg[2] = -cm * s;  gg[3] = -sm * s;   // m01 = -e^{+i(phi-om)/2} sin
        gg[4] =  cm * s;  gg[5] = -sm * s;   // m10 = e^{-i(phi-om)/2} sin
        gg[6] =  cp * c;  gg[7] =  sp * c;   // m11 = e^{+i(phi+om)/2} cos
    }

    sre[t]        = (t == j)        ? 1.0f : 0.0f;  sim_[t]        = 0.0f;
    sre[t + 512]  = (t + 512 == j)  ? 1.0f : 0.0f;  sim_[t + 512]  = 0.0f;
    __syncthreads();

    for (int l = 0; l < QDEPTH; ++l) {
        // Rot on every wire (pairs are disjoint per gate -> sync per gate only)
        for (int w = 0; w < WIRES; ++w) {
            const float* gg = g + (l * WIRES + w) * 8;
            const int p   = 9 - w;
            const int msk = 1 << p;
            const int i0  = ((t >> p) << (p + 1)) | (t & (msk - 1));
            const int i1  = i0 | msk;
            const float ar = sre[i0], ai = sim_[i0];
            const float br = sre[i1], bi = sim_[i1];
            const float nr0 = gg[0]*ar - gg[1]*ai + gg[2]*br - gg[3]*bi;
            const float ni0 = gg[0]*ai + gg[1]*ar + gg[2]*bi + gg[3]*br;
            const float nr1 = gg[4]*ar - gg[5]*ai + gg[6]*br - gg[7]*bi;
            const float ni1 = gg[4]*ai + gg[5]*ar + gg[6]*bi + gg[7]*br;
            sre[i0] = nr0; sim_[i0] = ni0;
            sre[i1] = nr1; sim_[i1] = ni1;
            __syncthreads();
        }
        // CNOT ring, range r (gates overlap wires -> sequential)
        const int r = (l % (WIRES - 1)) + 1;
        for (int w = 0; w < WIRES; ++w) {
            const int pc  = 9 - w;
            const int pt  = 9 - ((w + r) % WIRES);
            const int mskt = 1 << pt;
            const int i0  = ((t >> pt) << (pt + 1)) | (t & (mskt - 1));
            if ((i0 >> pc) & 1) {
                const int i1 = i0 | mskt;
                const float tr = sre[i0], ti = sim_[i0];
                sre[i0] = sre[i1]; sim_[i0] = sim_[i1];
                sre[i1] = tr;      sim_[i1] = ti;
            }
            __syncthreads();
        }
    }

    // Emit row K=j of B in fragment layout: h = (j>>3)&1, e = (j&7) + 8*((j>>4)&1),
    // fragment lane = (n&15) + 16h, tile = (j>>5, n>>4).
    const int kt = j >> 5;
    const int h  = (j >> 3) & 1;
    const int e  = (j & 7) + (((j >> 4) & 1) << 3);
    for (int i = t; i < PIXELS; i += 512) {
        const int nt = i >> 4;
        const int n  = i & 15;
        const size_t idx = (((size_t)kt * NTILES + nt) * 32 + (n + (h << 4))) * 16 + e;
        Bre[idx] = (_Float16)sre[i];
        Bim[idx] = (_Float16)sim_[i];
    }
}

// ---------------------------------------------------------------------------
// Kernel 3: WMMA GEMM (M=8192, N=784, K=1024) for real and imaginary parts,
// fused epilogue: out = clip((re^2+im^2)*784, 0, 1).  One wave per 16x16
// output tile; fragments are pre-swizzled so loads are contiguous 32B/lane.
// ---------------------------------------------------------------------------
__global__ __launch_bounds__(256) void k_gemm(const _Float16* __restrict__ Apack,
                                              const _Float16* __restrict__ Bre,
                                              const _Float16* __restrict__ Bim,
                                              float* __restrict__ out)
{
    const int wave = threadIdx.x >> 5;
    const int lane = threadIdx.x & 31;
    const int tile = blockIdx.x * 8 + wave;
    const int mt   = tile / NTILES;
    const int nt   = tile - mt * NTILES;

    const v16h* ap  = (const v16h*)(Apack) + (size_t)mt * KTILES * 32;
    const v16h* brp = (const v16h*)(Bre);
    const v16h* bip = (const v16h*)(Bim);

    v8f accR = {};
    v8f accI = {};
    for (int kt = 0; kt < KTILES; ++kt) {
        const v16h a  = ap[(size_t)kt * 32 + lane];
        const v16h br = brp[((size_t)kt * NTILES + nt) * 32 + lane];
        const v16h bi = bip[((size_t)kt * NTILES + nt) * 32 + lane];
        accR = __builtin_amdgcn_wmma_f32_16x16x32_f16(false, a, false, br,
                                                      (short)0, accR, false, false);
        accI = __builtin_amdgcn_wmma_f32_16x16x32_f16(false, a, false, bi,
                                                      (short)0, accI, false, false);
    }

    // C/D layout: lane = N + 16*(M>=8), VGPR r -> M = r + 8*(lane>>4)
    const int n     = nt * 16 + (lane & 15);
    const int mbase = mt * 16 + ((lane >> 4) << 3);
#pragma unroll
    for (int r = 0; r < 8; ++r) {
        const float re = accR[r];
        const float im = accI[r];
        float pr = (re * re + im * im) * (float)PIXELS;
        pr = fminf(fmaxf(pr, 0.0f), 1.0f);
        out[(size_t)(mbase + r) * PIXELS + n] = pr;
    }
}

// ---------------------------------------------------------------------------
extern "C" void kernel_launch(void* const* d_in, const int* in_sizes, int n_in,
                              void* d_out, int out_size, void* d_ws, size_t ws_size,
                              hipStream_t stream)
{
    (void)in_sizes; (void)n_in; (void)out_size; (void)ws_size;

    const float* x    = (const float*)d_in[0];   // 8192*784
    const float* W    = (const float*)d_in[1];   // 10*784
    const float* bias = (const float*)d_in[2];   // 10
    const float* qw   = (const float*)d_in[3];   // 20*10*3
    float* out        = (float*)d_out;           // 8192*784 f32

    char* ws = (char*)d_ws;
    _Float16* Apack = (_Float16*)ws;                               // 16,777,216 B
    _Float16* Bre   = (_Float16*)(ws + 16777216);                  //  1,605,632 B
    _Float16* Bim   = (_Float16*)(ws + 16777216 + 1605632);        //  1,605,632 B

    k_embed  <<<BATCH / 8, 256, 0, stream>>>(x, W, bias, Apack);
    k_unitary<<<DIM,       512, 0, stream>>>(qw, Bre, Bim);
    k_gemm   <<<(MTILES * NTILES) / 8, 256, 0, stream>>>(Apack, Bre, Bim, out);
}
